// StackedLSTMOne_36636071035709
// MI455X (gfx1250) — matrix operand order repaired and checked by reading, not compiled, not run on hardware
//
#include <hip/hip_runtime.h>

typedef __attribute__((ext_vector_type(16))) __bf16         v16bf;
typedef __attribute__((ext_vector_type(8)))  float          v8f;
typedef __attribute__((ext_vector_type(8)))  unsigned short v8u16;
typedef unsigned short u16;

#define BATCH 128
#define HID   1024
#define STEPS 64
#define XSTART 55          // T - NUM_STEPS - 9 = 128 - 64 - 9

// ---------- helpers ----------

__device__ __forceinline__ u16 f2bf(float f) {
  union { float f; unsigned u; } v; v.f = f;
  unsigned u = v.u;
  u += 0x7FFFu + ((u >> 16) & 1u);   // round-to-nearest-even
  return (u16)(u >> 16);
}

__device__ __forceinline__ float sigmoidf(float x) {
  return 1.0f / (1.0f + __expf(-x));
}

__device__ __forceinline__ v16bf mk_frag(v8u16 lo, v8u16 hi) {
  v16bf r;
#pragma unroll
  for (int j = 0; j < 8; ++j) {
    r[j]     = __builtin_bit_cast(__bf16, (unsigned short)lo[j]);
    r[j + 8] = __builtin_bit_cast(__bf16, (unsigned short)hi[j]);
  }
  return r;
}

// B fragment: 32x16 bf16 tile of W^T; W row-major [N, K] so K is contiguous
// per lane. Lanes 0-15 hold K=0..15, lanes 16-31 hold K=16..31.
__device__ __forceinline__ v16bf frag_b(const u16* __restrict__ W, int ldw,
                                        int n0, int k0, int lane) {
  const u16* p = W + (size_t)(n0 + (lane & 15)) * ldw + k0 + ((lane >> 4) << 4);
  return mk_frag(*reinterpret_cast<const v8u16*>(p),
                 *reinterpret_cast<const v8u16*>(p + 8));
}

// ---------- fused LSTM layer: gates GEMM + cell epilogue ----------
// grid = 128 blocks x 128 threads. Block -> one M-tile (16 rows), 4 j-tiles
// (one per wave). A-panel (16 x (K1+K2) bf16) staged in LDS once per block.
// Each wave holds i/f/g/o accumulators for its 16x16 (m,j) block and applies
// the sigmoid/tanh cell in-register, writing h/c/h_bf directly.
// B stream is software-pipelined: fragments for k resident while k+32 loads.

template <int K1, int K2>
__global__ void __launch_bounds__(128)
lstm_layer_fused(const u16* __restrict__ A1, const u16* __restrict__ W1,
                 const u16* __restrict__ A2, const u16* __restrict__ W2,
                 const float* __restrict__ b_ih, const float* __restrict__ b_hh,
                 float* __restrict__ cst, float* __restrict__ hst,
                 u16* __restrict__ hbf_out) {
  extern __shared__ u16 As[];                 // [16][K1+K2]
  constexpr int Ktot = K1 + K2;
  const int tid  = (int)threadIdx.x;
  const int lane = tid & 31;
  const int wave = tid >> 5;
  const int mt   = (int)blockIdx.x >> 4;      // 0..7
  const int jt   = ((int)blockIdx.x & 15) * 4 + wave;   // 0..63
  const int m0   = mt * 16;
  const int j0   = jt * 16;

  // cooperative stage: rows m0..m0+15 of A1 (cols [0,K1)) and A2 ([K1,Ktot))
  constexpr int Kv   = Ktot >> 3;             // v8u16 per row
  constexpr int vecs = 16 * Kv;
  for (int e = tid; e < vecs; e += 128) {
    const int row = e / Kv;
    const int k   = (e - row * Kv) << 3;
    v8u16 v;
    if (k < K1)
      v = *reinterpret_cast<const v8u16*>(A1 + (size_t)(m0 + row) * K1 + k);
    else
      v = *reinterpret_cast<const v8u16*>(A2 + (size_t)(m0 + row) * K2 + (k - K1));
    *reinterpret_cast<v8u16*>(&As[(size_t)row * Ktot + k]) = v;
  }
  __syncthreads();

  v8f zero = {};
  v8f acc[4];                                  // gates i, f, g, o
#pragma unroll
  for (int g = 0; g < 4; ++g) acc[g] = zero;

  // A frag from LDS: lanes 0-15 -> K [0..15]; 16-31 -> K [16..31]
  const int arow = lane & 15;
  const int akb  = (lane >> 4) << 4;

  auto panel = [&](const u16* __restrict__ W, const int K, const int kofs) {
    const u16* wp[4];
#pragma unroll
    for (int g = 0; g < 4; ++g)
      wp[g] = W + (size_t)(g * HID + j0 + (lane & 15)) * K + ((lane >> 4) << 4);
    v8u16 bl[4], bh[4];
#pragma unroll
    for (int g = 0; g < 4; ++g) {
      bl[g] = *reinterpret_cast<const v8u16*>(wp[g]);
      bh[g] = *reinterpret_cast<const v8u16*>(wp[g] + 8);
    }
    for (int k = 0; k < K; k += 32) {
      const u16* ap = &As[(size_t)arow * Ktot + kofs + k + akb];
      v16bf a = mk_frag(*reinterpret_cast<const v8u16*>(ap),
                        *reinterpret_cast<const v8u16*>(ap + 8));
      v8u16 nl[4], nh[4];
      const int kn = k + 32;
      if (kn < K) {
#pragma unroll
        for (int g = 0; g < 4; ++g) {
          nl[g] = *reinterpret_cast<const v8u16*>(wp[g] + kn);
          nh[g] = *reinterpret_cast<const v8u16*>(wp[g] + kn + 8);
        }
        __builtin_prefetch(wp[0] + kn + 32, 0, 1);
      } else {
#pragma unroll
        for (int g = 0; g < 4; ++g) { nl[g] = bl[g]; nh[g] = bh[g]; }
      }
#pragma unroll
      for (int g = 0; g < 4; ++g)
        acc[g] = __builtin_amdgcn_wmma_f32_16x16x32_bf16(
            false, a, false, mk_frag(bl[g], bh[g]), (short)0, acc[g],
            false, false);
#pragma unroll
      for (int g = 0; g < 4; ++g) { bl[g] = nl[g]; bh[g] = nh[g]; }
    }
  };
  panel(W1, K1, 0);
  panel(W2, K2, K1);

  // fused cell epilogue. D layout: VGPR r -> M = r + 8*(lane>>4), N = lane&15.
  const int n = j0 + (lane & 15);
  float bs[4];
#pragma unroll
  for (int g = 0; g < 4; ++g) bs[g] = b_ih[g * HID + n] + b_hh[g * HID + n];
  const int mb = m0 + ((lane >> 4) << 3);
#pragma unroll
  for (int r = 0; r < 8; ++r) {
    const size_t idx = (size_t)(mb + r) * HID + n;
    const float vi = sigmoidf(acc[0][r] + bs[0]);
    const float vf = sigmoidf(acc[1][r] + bs[1]);
    const float vg = tanhf(acc[2][r] + bs[2]);
    const float vo = sigmoidf(acc[3][r] + bs[3]);
    const float cn = vf * cst[idx] + vi * vg;
    const float hn = vo * tanhf(cn);
    cst[idx] = cn;
    hst[idx] = hn;
    hbf_out[idx] = f2bf(hn);
  }
}

// ---------- classifier GEMM: out[128,1000] = A @ W^T + bias ----------
// One wave -> 4 M-tiles x 1 N-tile (B reuse x4). N padded to 1008.

__global__ void __launch_bounds__(128)
cls_gemm(const u16* __restrict__ A, const u16* __restrict__ W,
         const float* __restrict__ bias, float* __restrict__ out) {
  const int lane = (int)threadIdx.x & 31;
  const int wid  = (int)((blockIdx.x * blockDim.x + threadIdx.x) >> 5);
  const int Ntiles = 63;
  if (wid >= 2 * Ntiles) return;
  const int n0 = (wid % Ntiles) * 16;
  const int m0 = (wid / Ntiles) * 64;

  v8f zero = {};
  v8f acc[4];
#pragma unroll
  for (int i = 0; i < 4; ++i) acc[i] = zero;

  for (int k = 0; k < HID; k += 32) {
    v16bf b = frag_b(W, HID, n0, k, lane);
    __builtin_prefetch(W + (size_t)(n0 + (lane & 15)) * HID + k + 32, 0, 1);
#pragma unroll
    for (int i = 0; i < 4; ++i) {
      const u16* p = A + (size_t)(m0 + 16 * i + (lane & 15)) * HID + k +
                     ((lane >> 4) << 4);
      v16bf a = mk_frag(*reinterpret_cast<const v8u16*>(p),
                        *reinterpret_cast<const v8u16*>(p + 8));
      acc[i] = __builtin_amdgcn_wmma_f32_16x16x32_bf16(
          false, a, false, b, (short)0, acc[i], false, false);
    }
  }
  const int n = n0 + (lane & 15);
  if (n >= 1000) return;
  const float bv = bias[n];
#pragma unroll
  for (int i = 0; i < 4; ++i) {
    const int mb = m0 + 16 * i + ((lane >> 4) << 3);
#pragma unroll
    for (int r = 0; r < 8; ++r)
      out[(size_t)(mb + r) * 1000 + n] = acc[i][r] + bv;
  }
}

// ---------- setup kernels ----------

__global__ void cvt_f32_bf16(const float* __restrict__ s, u16* __restrict__ d, int n) {
  const int i = blockIdx.x * blockDim.x + threadIdx.x;
  if (i < n) d[i] = f2bf(s[i]);
}

__global__ void cvt_cls_pad(const float* __restrict__ s, u16* __restrict__ d) {
  const int i = blockIdx.x * blockDim.x + threadIdx.x;   // [1008][1024]
  if (i >= 1008 * 1024) return;
  const int n = i >> 10, k = i & 1023;
  d[i] = (n < 1000) ? f2bf(s[(size_t)n * 1024 + k]) : (u16)0;
}

__global__ void slice_xs_bf16(const float* __restrict__ xs, u16* __restrict__ xb) {
  const int i = blockIdx.x * blockDim.x + threadIdx.x;   // [64][128][512]
  if (i >= STEPS * BATCH * 512) return;
  const int t   = i / (BATCH * 512);
  const int rem = i - t * (BATCH * 512);
  const int b   = rem >> 9;
  const int dd  = rem & 511;
  xb[i] = f2bf(xs[(size_t)b * (128 * 512) + (size_t)(XSTART + t) * 512 + dd]);
}

__global__ void zero_words(unsigned* __restrict__ p, int n) {
  const int i = blockIdx.x * blockDim.x + threadIdx.x;
  if (i < n) p[i] = 0u;
}

__global__ void dropout_bf16(const float* __restrict__ h1,
                             const float* __restrict__ u,
                             u16* __restrict__ a) {
  const int i = blockIdx.x * blockDim.x + threadIdx.x;
  if (i >= BATCH * HID) return;
  const float keep = (u[i] > 0.5f) ? 2.0f : 0.0f;   // /(1-p) folded in
  a[i] = f2bf(h1[i] * keep);
}

// ---------- launch ----------

extern "C" void kernel_launch(void* const* d_in, const int* in_sizes, int n_in,
                              void* d_out, int out_size, void* d_ws, size_t ws_size,
                              hipStream_t stream) {
  (void)in_sizes; (void)n_in; (void)out_size; (void)ws_size;
  const float* xs    = (const float*)d_in[0];
  const float* W_ih0 = (const float*)d_in[1];
  const float* W_hh0 = (const float*)d_in[2];
  const float* b_ih0 = (const float*)d_in[3];
  const float* b_hh0 = (const float*)d_in[4];
  const float* W_ih1 = (const float*)d_in[5];
  const float* W_hh1 = (const float*)d_in[6];
  const float* b_ih1 = (const float*)d_in[7];
  const float* b_hh1 = (const float*)d_in[8];
  const float* W_cls = (const float*)d_in[9];
  const float* b_cls = (const float*)d_in[10];
  const float* dropu = (const float*)d_in[11];
  float* out = (float*)d_out;

  char* ws = (char*)d_ws;
  size_t off = 0;
  auto take = [&](size_t bytes) -> char* {
    char* p = ws + off;
    off += (bytes + 255) & ~(size_t)255;
    return p;
  };

  u16* W_ih0_bf = (u16*)take((size_t)4096 * 512  * 2);
  u16* W_hh0_bf = (u16*)take((size_t)4096 * 1024 * 2);
  u16* W_ih1_bf = (u16*)take((size_t)4096 * 1024 * 2);
  u16* W_hh1_bf = (u16*)take((size_t)4096 * 1024 * 2);
  u16* W_cls_bf = (u16*)take((size_t)1008 * 1024 * 2);
  u16* x_bf     = (u16*)take((size_t)STEPS * BATCH * 512 * 2);
  float* c0 = (float*)take((size_t)BATCH * HID * 4);    // c0,c1 contiguous
  float* c1 = (float*)take((size_t)BATCH * HID * 4);
  float* h0 = (float*)take((size_t)BATCH * HID * 4);
  float* h1 = (float*)take((size_t)BATCH * HID * 4);
  u16* hbf0[2]; u16* hbf1[2];                           // 4 contiguous buffers
  hbf0[0] = (u16*)take((size_t)BATCH * HID * 2);
  hbf0[1] = (u16*)take((size_t)BATCH * HID * 2);
  hbf1[0] = (u16*)take((size_t)BATCH * HID * 2);
  hbf1[1] = (u16*)take((size_t)BATCH * HID * 2);
  u16* a_cls = (u16*)take((size_t)BATCH * HID * 2);
  (void)c1;

  const int TPB = 256;
  auto blks = [](int n, int t) { return (n + t - 1) / t; };

  // per-call conversions; bf16 weights (~28 MB) stay resident in 192 MB L2
  cvt_f32_bf16<<<blks(4096 * 512, TPB),  TPB, 0, stream>>>(W_ih0, W_ih0_bf, 4096 * 512);
  cvt_f32_bf16<<<blks(4096 * 1024, TPB), TPB, 0, stream>>>(W_hh0, W_hh0_bf, 4096 * 1024);
  cvt_f32_bf16<<<blks(4096 * 1024, TPB), TPB, 0, stream>>>(W_ih1, W_ih1_bf, 4096 * 1024);
  cvt_f32_bf16<<<blks(4096 * 1024, TPB), TPB, 0, stream>>>(W_hh1, W_hh1_bf, 4096 * 1024);
  cvt_cls_pad<<<blks(1008 * 1024, TPB),  TPB, 0, stream>>>(W_cls, W_cls_bf);
  slice_xs_bf16<<<blks(STEPS * BATCH * 512, TPB), TPB, 0, stream>>>(xs, x_bf);

  // zero c0,c1 (contiguous fp32) and all four bf16 h ping-pong buffers
  zero_words<<<blks(2 * BATCH * HID, TPB), TPB, 0, stream>>>((unsigned*)c0, 2 * BATCH * HID);
  zero_words<<<blks(2 * BATCH * HID, TPB), TPB, 0, stream>>>((unsigned*)hbf0[0], 2 * BATCH * HID);

  // recurrent chain: 2 fused launches per step; stream order is the dependency
  const size_t lds0 = (size_t)16 * (512 + HID) * 2;     // 48 KB
  const size_t lds1 = (size_t)16 * (HID + HID) * 2;     // 64 KB (<= 320 KB/WGP)
  for (int t = 0; t < STEPS; ++t) {
    const int ri = t & 1, wi = ri ^ 1;
    lstm_layer_fused<512, 1024><<<128, 128, lds0, stream>>>(
        x_bf + (size_t)t * BATCH * 512, W_ih0_bf,
        hbf0[ri], W_hh0_bf, b_ih0, b_hh0, c0, h0, hbf0[wi]);
    lstm_layer_fused<1024, 1024><<<128, 128, lds1, stream>>>(
        hbf0[wi], W_ih1_bf,
        hbf1[ri], W_hh1_bf, b_ih1, b_hh1, c1, h1, hbf1[wi]);
  }

  // dropout + classifier
  dropout_bf16<<<blks(BATCH * HID, TPB), TPB, 0, stream>>>(h1, dropu, a_cls);
  cls_gemm<<<32, 128, 0, stream>>>(a_cls, W_cls_bf, b_cls, out);
}
